// NetMambaCore_2714419331858
// MI455X (gfx1250) — matrix operand order repaired and checked by reading, not compiled, not run on hardware
//
#include <hip/hip_runtime.h>

// ---------------------------------------------------------------------------
// Mamba forward (4 layers) for MI455X / gfx1250 (wave32, WMMA).
// GEMMs: v_wmma_f32_16x16x32_f16, compile-time K fully unrolled so the
//        scheduler pipelines loads under WMMAs without register copies.
// Scan: f32 VALU recurrence, state in VGPRs, B/C staged via
//       global_load_async_to_lds_b32 (ASYNCcnt path).
// ---------------------------------------------------------------------------

#define D_MODEL  128
#define D_INNER  256
#define D_STATE  16
#define DT_RANK  8
#define LSEQ     4096
#define BATCH    8
#define MROWS    (BATCH * LSEQ)     // 32768 token rows
#define PROJ_LD  48                 // x_proj output padded 40 -> 48

typedef __attribute__((ext_vector_type(16))) _Float16 v16h;
typedef __attribute__((ext_vector_type(8)))  _Float16 v8h;
typedef __attribute__((ext_vector_type(8)))  float    v8f;

// ---------------------------------------------------------------------------
// Generic f32 -> f16 convert
// ---------------------------------------------------------------------------
__global__ void cvt_f16_kernel(const float* __restrict__ src,
                               _Float16* __restrict__ dst, int n) {
  int i = blockIdx.x * blockDim.x + threadIdx.x;
  if (i < n) dst[i] = (_Float16)src[i];
}

// x_proj weights [4][40][256] f32 -> [4][48][256] f16, zero-padded rows 40..47
__global__ void cvt_pad_xpw_kernel(const float* __restrict__ src,
                                   _Float16* __restrict__ dst) {
  int i = blockIdx.x * blockDim.x + threadIdx.x;   // 4*48*256 total
  int layer = i / (PROJ_LD * D_INNER);
  int rem   = i - layer * (PROJ_LD * D_INNER);
  int r = rem >> 8;          // 0..47
  int c = rem & 255;         // 0..255
  float v = (r < (DT_RANK + 2 * D_STATE))
              ? src[(size_t)layer * 40 * D_INNER + r * D_INNER + c] : 0.0f;
  dst[i] = (_Float16)v;
}

// ---------------------------------------------------------------------------
// Fragment loads per CDNA5 ISA 7.12.2 (16-bit A 16x32; B 32x16).
// Caller applies the lane-dependent base offsets.
// ---------------------------------------------------------------------------
__device__ __forceinline__ v16h ld_fragA(const _Float16* p) {
  v8h lo = *(const v8h*)p;        // K = koffA .. +7
  v8h hi = *(const v8h*)(p + 16); // K = 16+koffA .. +7
  v16h r;
#pragma unroll
  for (int i = 0; i < 8; ++i) { r[i] = lo[i]; r[i + 8] = hi[i]; }
  return r;
}
__device__ __forceinline__ v16h ld_fragB(const _Float16* p) {
  v8h lo = *(const v8h*)p;        // K = koffB .. +7
  v8h hi = *(const v8h*)(p + 16); // K = 16+koffB .. +7
  v16h r;
#pragma unroll
  for (int i = 0; i < 8; ++i) { r[i] = lo[i]; r[i + 8] = hi[i]; }
  return r;
}

// ---------------------------------------------------------------------------
// WMMA GEMM: C[M,N] = A[M,K] * W[N,K]^T   (A,W f16 row-major, C f32)
// One wave computes a 16 x (16*NT) tile; A-fragment reused NT times.
// N, K compile-time: K-loop fully unrolls, scheduler pipelines the loads.
// ---------------------------------------------------------------------------
template <int NT, int N, int K>
__global__ void __launch_bounds__(128)
gemm_f16_wmma(const _Float16* __restrict__ A, const _Float16* __restrict__ W,
              float* __restrict__ C) {
  const int lane = threadIdx.x & 31;
  const int wave = blockIdx.x * 4 + (threadIdx.x >> 5);
  constexpr int nGroups = N / (16 * NT);
  const int m0 = (wave / nGroups) * 16;
  const int n0 = (wave % nGroups) * (16 * NT);

  v8f acc[NT];
#pragma unroll
  for (int j = 0; j < NT; ++j) acc[j] = (v8f){};

  const _Float16* pA0 =
      A + (size_t)(m0 + (lane & 15)) * K + ((lane >> 4) << 3);
  const _Float16* pB0 =
      W + (size_t)(n0 + (lane & 15)) * K + ((lane >> 4) << 3);

#pragma unroll
  for (int kb = 0; kb < K; kb += 32) {
    v16h af = ld_fragA(pA0 + kb);
#pragma unroll
    for (int j = 0; j < NT; ++j) {
      v16h bf = ld_fragB(pB0 + (size_t)j * 16 * K + kb);
      acc[j] = __builtin_amdgcn_wmma_f32_16x16x32_f16(
          false, af, false, bf, (short)0, acc[j], false, false);
    }
  }

  // D layout: VGPR r holds M = m0 + r + 8*(lane>=16), N = n0 + (lane&15)
  const int mbase = m0 + ((lane >> 4) << 3);
  const int nl    = lane & 15;
#pragma unroll
  for (int j = 0; j < NT; ++j)
#pragma unroll
    for (int r = 0; r < 8; ++r)
      C[(size_t)(mbase + r) * N + n0 + j * 16 + nl] = acc[j][r];
}

// ---------------------------------------------------------------------------
// Causal depthwise conv (k=4) + SiLU on xa = xz[:, 0:256].
// Writes f32 (for scan input u) and f16 (for x_proj GEMM).
// ---------------------------------------------------------------------------
__global__ void conv_silu_kernel(const float* __restrict__ xz,
                                 const float* __restrict__ cw,
                                 const float* __restrict__ cb,
                                 float* __restrict__ xc,
                                 _Float16* __restrict__ xch) {
  int idx = blockIdx.x * blockDim.x + threadIdx.x;  // MROWS*256
  int d = idx & 255;
  int m = idx >> 8;
  int l = m & (LSEQ - 1);
  float v = cb[d];
#pragma unroll
  for (int k = 0; k < 4; ++k) {
    int ls = l + k - 3;
    if (ls >= 0) v += cw[d * 4 + k] * xz[(size_t)(m + k - 3) * 512 + d];
  }
  float s = v / (1.0f + __expf(-v));   // SiLU
  xc[(size_t)m * 256 + d] = s;
  xch[(size_t)m * 256 + d] = (_Float16)s;
}

// ---------------------------------------------------------------------------
// dt = softplus(proj[:, 0:8] @ dt_proj_w^T + dt_proj_b) -> stored into the
// (now dead) xa columns of xz (stride 512), keeping z columns intact.
// ---------------------------------------------------------------------------
__global__ void dt_softplus_kernel(const float* __restrict__ proj,
                                   const float* __restrict__ dtw,
                                   const float* __restrict__ dtb,
                                   float* __restrict__ dt_out) {
  int idx = blockIdx.x * blockDim.x + threadIdx.x;  // MROWS*256
  int d = idx & 255;
  int m = idx >> 8;
  float s = dtb[d];
#pragma unroll
  for (int r = 0; r < DT_RANK; ++r)
    s += proj[(size_t)m * PROJ_LD + r] * dtw[d * DT_RANK + r];
  float sp = (s > 20.0f) ? s : log1pf(__expf(s));
  dt_out[(size_t)m * 512 + d] = sp;
}

// ---------------------------------------------------------------------------
// Selective scan, fused with D-skip and SiLU(z) gating. One wave per
// (batch, 32-channel chunk): lane owns channel d, 16-entry state in VGPRs.
// B/C vectors staged through LDS via global_load_async_to_lds_b32
// (ASYNCcnt), in chunks of 8 timesteps.
// ---------------------------------------------------------------------------
__global__ void __launch_bounds__(32)
scan_kernel(const float* __restrict__ xz,     // dt in cols 0..255, z in 256..511
            const float* __restrict__ xc,     // u  [M,256]
            const float* __restrict__ proj,   // B,C at cols 8..39, ld=48
            const float* __restrict__ A_log,  // [256,16]
            const float* __restrict__ Dvec,   // [256]
            _Float16* __restrict__ yh) {      // [M,256] f16
  const int b    = blockIdx.x >> 3;
  const int dc   = blockIdx.x & 7;
  const int lane = threadIdx.x;               // 0..31
  const int d    = dc * 32 + lane;

  float a[D_STATE], h[D_STATE];
#pragma unroll
  for (int n = 0; n < D_STATE; ++n) {
    a[n] = -__expf(A_log[d * D_STATE + n]);
    h[n] = 0.0f;
  }
  const float Dp = Dvec[d];

  __shared__ float sBC[8][32];                // [step][B(0..15) | C(16..31)]
  // Escape the LDS base so the async-to-LDS asm below is seen as a writer.
  asm volatile("" :: "v"(&sBC[0][0]) : "memory");

  const int base = b * LSEQ;

  for (int t0 = 0; t0 < LSEQ; t0 += 8) {
    // WAR: previous chunk's ds_load reads must retire before overwrite.
    asm volatile("s_wait_dscnt 0" ::: "memory");
    __syncthreads();
#pragma unroll
    for (int s = 0; s < 8; ++s) {
      const float* g =
          proj + (size_t)(base + t0 + s) * PROJ_LD + DT_RANK + lane;
      unsigned ldsOff = (unsigned)(s * 32 + lane) * 4u;  // sBC starts at LDS 0
      asm volatile("global_load_async_to_lds_b32 %0, %1, off"
                   :: "v"(ldsOff), "v"((unsigned long long)(uintptr_t)g)
                   : "memory");
    }
    asm volatile("s_wait_asynccnt 0" ::: "memory");
    __syncthreads();

#pragma unroll
    for (int s = 0; s < 8; ++s) {
      const int m = base + t0 + s;
      const float dtv = xz[(size_t)m * 512 + d];
      const float u   = xc[(size_t)m * 256 + d];
      const float zv  = xz[(size_t)m * 512 + 256 + d];
      const float du  = dtv * u;
      float acc = 0.0f;
#pragma unroll
      for (int n = 0; n < D_STATE; ++n) {
        float dA = __expf(dtv * a[n]);        // v_exp_f32 (TRANS, co-executes)
        h[n] = dA * h[n] + du * sBC[s][n];
        acc += h[n] * sBC[s][16 + n];
      }
      float y = acc + u * Dp;
      y *= zv / (1.0f + __expf(-zv));         // * SiLU(z)
      yh[(size_t)m * 256 + d] = (_Float16)y;
    }
  }
}

// ---------------------------------------------------------------------------
// Host launcher
// ---------------------------------------------------------------------------
extern "C" void kernel_launch(void* const* d_in, const int* in_sizes, int n_in,
                              void* d_out, int out_size, void* d_ws, size_t ws_size,
                              hipStream_t stream) {
  (void)in_sizes; (void)n_in; (void)out_size; (void)ws_size;

  const float* x     = (const float*)d_in[0];
  const float* inw   = (const float*)d_in[1];
  const float* convw = (const float*)d_in[2];
  const float* convb = (const float*)d_in[3];
  const float* xpw   = (const float*)d_in[4];
  const float* dtw   = (const float*)d_in[5];
  const float* dtb   = (const float*)d_in[6];
  const float* A_log = (const float*)d_in[7];
  const float* Dvec  = (const float*)d_in[8];
  const float* outw  = (const float*)d_in[9];
  float* out = (float*)d_out;

  char* p = (char*)d_ws;
  auto alloc = [&](size_t bytes) -> char* {
    char* r = p;
    p += (bytes + 255) & ~(size_t)255;
    return r;
  };
  _Float16* xh   = (_Float16*)alloc((size_t)MROWS * 128 * 2);
  float*    xz   = (float*)   alloc((size_t)MROWS * 512 * 4);
  float*    xc   = (float*)   alloc((size_t)MROWS * 256 * 4);
  _Float16* xch  = (_Float16*)alloc((size_t)MROWS * 256 * 2);
  float*    proj = (float*)   alloc((size_t)MROWS * PROJ_LD * 4);
  _Float16* yh   = (_Float16*)alloc((size_t)MROWS * 256 * 2);
  float*    xbuf = (float*)   alloc((size_t)MROWS * 128 * 4);
  _Float16* wih  = (_Float16*)alloc((size_t)4 * 512 * 128 * 2);
  _Float16* wph  = (_Float16*)alloc((size_t)4 * PROJ_LD * 256 * 2);
  _Float16* woh  = (_Float16*)alloc((size_t)4 * 128 * 256 * 2);

  // Weight conversions (constant per call; recomputed for determinism).
  cvt_f16_kernel<<<(4 * 512 * 128) / 256, 256, 0, stream>>>(inw, wih, 4 * 512 * 128);
  cvt_pad_xpw_kernel<<<(4 * PROJ_LD * 256) / 256, 256, 0, stream>>>(xpw, wph);
  cvt_f16_kernel<<<(4 * 128 * 256) / 256, 256, 0, stream>>>(outw, woh, 4 * 128 * 256);

  for (int layer = 0; layer < 4; ++layer) {
    const float* xin = (layer == 0) ? x : xbuf;
    cvt_f16_kernel<<<(MROWS * 128) / 256, 256, 0, stream>>>(xin, xh, MROWS * 128);

    // in_proj: [M,128] x [512,128]^T -> xz [M,512]
    gemm_f16_wmma<4, 512, 128>
        <<<((MROWS / 16) * (512 / 64)) / 4, 128, 0, stream>>>(
            xh, wih + (size_t)layer * 512 * 128, xz);

    conv_silu_kernel<<<(MROWS * 256) / 256, 256, 0, stream>>>(
        xz, convw + (size_t)layer * 256 * 4, convb + (size_t)layer * 256, xc, xch);

    // x_proj: [M,256] x [48,256]^T -> proj [M,48]
    gemm_f16_wmma<1, PROJ_LD, 256>
        <<<((MROWS / 16) * (PROJ_LD / 16)) / 4, 128, 0, stream>>>(
            xch, wph + (size_t)layer * PROJ_LD * 256, proj);

    dt_softplus_kernel<<<(MROWS * 256) / 256, 256, 0, stream>>>(
        proj, dtw + (size_t)layer * 256 * DT_RANK, dtb + (size_t)layer * 256, xz);

    scan_kernel<<<BATCH * 8, 32, 0, stream>>>(
        xz, xc, proj, A_log + (size_t)layer * 256 * D_STATE,
        Dvec + (size_t)layer * 256, yh);

    // out_proj: [M,256] x [128,256]^T -> next x (or final output)
    float* yout = (layer == 3) ? out : xbuf;
    gemm_f16_wmma<4, 128, 256>
        <<<((MROWS / 16) * (128 / 64)) / 4, 128, 0, stream>>>(
            yh, woh + (size_t)layer * 128 * 256, yout);
  }
}
